// GATv2Layer_77738908057664
// MI455X (gfx1250) — matrix operand order repaired and checked by reading, not compiled, not run on hardware
//
#include <hip/hip_runtime.h>
#include <hip/hip_bf16.h>

// ---------------------------------------------------------------------------
// GATv2-style layer for MI455X (gfx1250, wave32).
//   x_lr[N,256] = nodes[N,128] @ [W_l | W_r]   (bf16 WMMA, f32 accum)
//   per-edge logits + segment softmax + weighted scatter-add + head mean
// Weights are pre-transposed to column-major so both A and B WMMA fragments
// load as contiguous 16-byte global_load_b128s.
// ---------------------------------------------------------------------------

typedef __attribute__((ext_vector_type(16))) __bf16 v16bf;
typedef __attribute__((ext_vector_type(8)))  __bf16 v8bf;
typedef __attribute__((ext_vector_type(8)))  float  v8f;

__device__ __forceinline__ __bf16 f2bf(float x) {
    // round-to-nearest-even f32 -> bf16
    unsigned u = __float_as_uint(x);
    unsigned r = (u + 0x7FFFu + ((u >> 16) & 1u)) >> 16;
    unsigned short h = (unsigned short)r;
    __bf16 o;
    __builtin_memcpy(&o, &h, 2);
    return o;
}

__device__ __forceinline__ void atomic_add_f32(float* a, float v) {
    // GLOBAL_ATOMIC_ADD_F32, no return (STOREcnt); implicit wait at s_endpgm.
    asm volatile("global_atomic_add_f32 %0, %1, off" :: "v"(a), "v"(v) : "memory");
}

__device__ __forceinline__ v16bf cat16(v8bf lo, v8bf hi) {
    return __builtin_shufflevector(lo, hi, 0, 1, 2, 3, 4, 5, 6, 7, 8, 9, 10,
                                   11, 12, 13, 14, 15);
}

// --- prep: bf16 conversions + accumulator init (runs every call) -----------
// WlrT is COLUMN-major: WlrT[c*128 + k] = (c<128 ? W_l[k][c] : W_r[k][c-128])
__global__ void __launch_bounds__(256)
k_prep(const float* __restrict__ nodes, const float* __restrict__ W_l,
       const float* __restrict__ W_r, __bf16* __restrict__ nodes_bf,
       __bf16* __restrict__ WlrT_bf, float* __restrict__ agg,
       float* __restrict__ amax, float* __restrict__ asum, int N) {
    int i = blockIdx.x * 256 + threadIdx.x;
    int totalNode = N * 128;
    if (i < totalNode) {
        nodes_bf[i] = f2bf(nodes[i]);
        agg[i] = 0.0f;
    }
    if (i < 128 * 256) {
        int k = i >> 8, c = i & 255;  // k = F_IN row, c = output column
        float w = (c < 128) ? W_l[k * 128 + c] : W_r[k * 128 + (c - 128)];
        WlrT_bf[(size_t)c * 128 + k] = f2bf(w);
    }
    if (i < N * 4) {
        amax[i] = -3.0e38f;
        asum[i] = 0.0f;
    }
}

// --- WMMA GEMM: C[N,256] = A[N,128] @ W[128,256] ---------------------------
// One wave computes a 16x64 output strip (4 C tiles), A fragment reused 4x.
// WT is column-major [256][128] so B fragments are contiguous along K.
__global__ void __launch_bounds__(32)
k_gemm(const __bf16* __restrict__ A, const __bf16* __restrict__ WT,
       float* __restrict__ C, int N) {
    const int m0 = blockIdx.x * 16;
    const int n0 = blockIdx.y * 64;
    const int lane = threadIdx.x;
    const int half = lane & 15;   // A: row M ; B/C: col N
    const int g    = lane >> 4;   // lane group
    int mrow = m0 + half;
    if (mrow >= N) mrow = N - 1;  // clamp loads; stores guarded below
    const __bf16* Arow = A + (size_t)mrow * 128;

    v8f acc[4] = {{}, {}, {}, {}};
#pragma unroll
    for (int k0 = 0; k0 < 128; k0 += 32) {
        // A 16x32 bf16: elems 0..7 -> K = k0+g*8+i ; 8..15 -> K = k0+16+g*8+i
        v8bf a0 = *(const v8bf*)(Arow + k0 + g * 8);
        v8bf a1 = *(const v8bf*)(Arow + k0 + 16 + g * 8);
        v16bf a = cat16(a0, a1);
#pragma unroll
        for (int t = 0; t < 4; ++t) {
            // B 32x16 bf16: elems 0..7 -> K = k0+g*8+i ; 8..15 -> K = k0+16+g*8+i, col n = n0 + t*16 + half
            const __bf16* Bcol = WT + (size_t)(n0 + t * 16 + half) * 128 + k0 + g * 8;
            v8bf b0 = *(const v8bf*)(Bcol);
            v8bf b1 = *(const v8bf*)(Bcol + 16);
            acc[t] = __builtin_amdgcn_wmma_f32_16x16x32_bf16(
                false, a, false, cat16(b0, b1), (short)0, acc[t], false, false);
        }
    }
    // C/D f32 16x16: VGPR r -> M = g*8 + r, col = n0 + t*16 + half
#pragma unroll
    for (int t = 0; t < 4; ++t) {
#pragma unroll
        for (int r = 0; r < 8; ++r) {
            int m = m0 + g * 8 + r;
            if (m < N) C[(size_t)m * 256 + n0 + t * 16 + half] = acc[t][r];
        }
    }
}

// --- edge logits + segment max: one wave per edge --------------------------
__global__ void __launch_bounds__(256)
k_edge_logits(const float* __restrict__ x_lr, const int* __restrict__ senders,
              const int* __restrict__ receivers, const float* __restrict__ edge_attr,
              const float* __restrict__ W_e, const float* __restrict__ attn_vec,
              float* __restrict__ logits, float* __restrict__ amax, int E) {
    __shared__ float sWe[16 * 128];
    __shared__ float sAv[128];
    for (int i = threadIdx.x; i < 16 * 128; i += 256) sWe[i] = W_e[i];
    if (threadIdx.x < 128) sAv[threadIdx.x] = attn_vec[threadIdx.x];
    __syncthreads();

    const int lane = threadIdx.x & 31;
    const int e = blockIdx.x * 8 + (threadIdx.x >> 5);
    if (e >= E) return;  // uniform per wave

    const int s = senders[e];
    const int r = receivers[e];
    float my = (lane < 16) ? edge_attr[(size_t)e * 16 + lane] : 0.0f;
    float ea[16];
#pragma unroll
    for (int j = 0; j < 16; ++j) ea[j] = __shfl(my, j, 32);

    const float* xls = x_lr + (size_t)s * 256;        // x_l row
    const float* xrr = x_lr + (size_t)r * 256 + 128;  // x_r row

#pragma unroll
    for (int h = 0; h < 4; ++h) {
        const int c = h * 32 + lane;
        float ep = 0.0f;
#pragma unroll
        for (int j = 0; j < 16; ++j) ep = fmaf(ea[j], sWe[j * 128 + c], ep);
        float m = xls[c] + xrr[c] + ep;
        m = (m > 0.0f) ? m : 0.2f * m;  // leaky_relu(0.2)
        float p = m * sAv[c];
#pragma unroll
        for (int off = 16; off > 0; off >>= 1) p += __shfl_xor(p, off, 32);
        if (lane == 0) {
            logits[(size_t)e * 4 + h] = p;
            float* addr = amax + (size_t)r * 4 + h;  // float max via int punning
            if (p >= 0.0f)
                atomicMax((int*)addr, __float_as_int(p));
            else
                atomicMin((unsigned int*)addr, __float_as_uint(p));
        }
    }
}

// --- exp(logit - max) + segment sum ----------------------------------------
__global__ void __launch_bounds__(256)
k_edge_exp(float* __restrict__ logits, const int* __restrict__ receivers,
           const float* __restrict__ amax, float* __restrict__ asum, int E4) {
    int i = blockIdx.x * 256 + threadIdx.x;
    if (i >= E4) return;
    int e = i >> 2, h = i & 3;
    int r = receivers[e];
    float ex = __expf(logits[i] - amax[(size_t)r * 4 + h]);
    logits[i] = ex;  // overwrite in place (same thread R/W)
    atomic_add_f32(asum + (size_t)r * 4 + h, ex);
}

// --- normalized weighted scatter-add: one wave per edge --------------------
__global__ void __launch_bounds__(256)
k_scatter(const float* __restrict__ x_lr, const float* __restrict__ logits,
          const float* __restrict__ asum, const int* __restrict__ senders,
          const int* __restrict__ receivers, float* __restrict__ agg, int E) {
    const int lane = threadIdx.x & 31;
    const int e = blockIdx.x * 8 + (threadIdx.x >> 5);
    if (e >= E) return;
    const int s = senders[e];
    const int r = receivers[e];
    const float* xls = x_lr + (size_t)s * 256;
    __builtin_prefetch(xls, 0, 0);  // global_prefetch_b8 on gfx1250
    float w[4];
#pragma unroll
    for (int h = 0; h < 4; ++h)
        w[h] = logits[(size_t)e * 4 + h] / (asum[(size_t)r * 4 + h] + 1e-8f);
    float* aggr = agg + (size_t)r * 128;
#pragma unroll
    for (int h = 0; h < 4; ++h) {
        int c = h * 32 + lane;
        atomic_add_f32(aggr + c, xls[c] * w[h]);
    }
}

// --- head mean: out[n,d] = 0.25 * sum_h agg[n, h*32+d] ---------------------
__global__ void __launch_bounds__(256)
k_final(const float* __restrict__ agg, float* __restrict__ out, int N32) {
    int i = blockIdx.x * 256 + threadIdx.x;
    if (i >= N32) return;
    int n = i >> 5, d = i & 31;
    const float* a = agg + (size_t)n * 128 + d;
    out[i] = 0.25f * (a[0] + a[32] + a[64] + a[96]);
}

// ---------------------------------------------------------------------------
extern "C" void kernel_launch(void* const* d_in, const int* in_sizes, int n_in,
                              void* d_out, int out_size, void* d_ws, size_t ws_size,
                              hipStream_t stream) {
    const float* nodes     = (const float*)d_in[0];
    const int*   senders   = (const int*)d_in[1];
    const int*   receivers = (const int*)d_in[2];
    const float* edge_attr = (const float*)d_in[3];
    // d_in[4] = n_node scalar (unused; derived from in_sizes)
    const float* W_l      = (const float*)d_in[5];
    const float* W_r      = (const float*)d_in[6];
    const float* W_e      = (const float*)d_in[7];
    const float* attn_vec = (const float*)d_in[8];
    float* out = (float*)d_out;

    const int N = in_sizes[0] / 128;
    const int E = in_sizes[1];

    // workspace carve-up (256 B aligned)
    char* ws = (char*)d_ws;
    size_t off = 0;
    auto alloc = [&](size_t bytes) {
        char* p = ws + off;
        off = (off + bytes + 255) & ~(size_t)255;
        return p;
    };
    float*  x_lr     = (float*)alloc((size_t)N * 256 * 4);
    float*  agg      = (float*)alloc((size_t)N * 128 * 4);
    float*  logits   = (float*)alloc((size_t)E * 4 * 4);
    float*  amax     = (float*)alloc((size_t)N * 4 * 4);
    float*  asum     = (float*)alloc((size_t)N * 4 * 4);
    __bf16* nodes_bf = (__bf16*)alloc((size_t)N * 128 * 2);
    __bf16* WlrT_bf  = (__bf16*)alloc((size_t)128 * 256 * 2);
    (void)ws_size;

    // 1) convert + init
    {
        int total = N * 128;
        k_prep<<<(total + 255) / 256, 256, 0, stream>>>(
            nodes, W_l, W_r, nodes_bf, WlrT_bf, agg, amax, asum, N);
    }
    // 2) fused projection GEMM (bf16 WMMA, 16x64 strip per wave)
    {
        dim3 grid((N + 15) / 16, 4);
        k_gemm<<<grid, 32, 0, stream>>>(nodes_bf, WlrT_bf, x_lr, N);
    }
    // 3) edge logits + segment max
    k_edge_logits<<<(E + 7) / 8, 256, 0, stream>>>(
        x_lr, senders, receivers, edge_attr, W_e, attn_vec, logits, amax, E);
    // 4) exp + segment sum
    k_edge_exp<<<(E * 4 + 255) / 256, 256, 0, stream>>>(
        logits, receivers, amax, asum, E * 4);
    // 5) weighted scatter-add
    k_scatter<<<(E + 7) / 8, 256, 0, stream>>>(
        x_lr, logits, asum, senders, receivers, agg, E);
    // 6) head mean
    k_final<<<(N * 32 + 255) / 256, 256, 0, stream>>>(agg, out, N * 32);
}
